// GCN_46634754900269
// MI455X (gfx1250) — matrix operand order repaired and checked by reading, not compiled, not run on hardware
//
#include <hip/hip_runtime.h>
#include <hip/hip_bf16.h>

// ---------------------------------------------------------------------------
// GCN forward for MI455X (gfx1250, wave32, WMMA).
//   support1 = x @ W1^T + b1        (bf16 WMMA, f32 accum)
//   h        = relu(spmm(support1)) (f32 atomics, L2-resident)
//   support2 = h @ W2^T + b2        (bf16 WMMA)
//   out      = spmm(support2)
// ---------------------------------------------------------------------------

typedef __attribute__((ext_vector_type(16))) __bf16 v16bf;
typedef __attribute__((ext_vector_type(8)))  __bf16 v8bf;
typedef __attribute__((ext_vector_type(8)))  float  v8f;

#define GCN_N      50000
#define GCN_E      800000
#define GCN_NFEAT  512
#define GCN_NHID   512
#define GCN_NCLASS 128

// ---------------------------------------------------------------------------
// f32 -> bf16 conversion (grid-stride)
// ---------------------------------------------------------------------------
__global__ __launch_bounds__(256) void cvt_f32_bf16(const float* __restrict__ in,
                                                    __bf16* __restrict__ out,
                                                    size_t n) {
  size_t i = (size_t)blockIdx.x * blockDim.x + threadIdx.x;
  size_t stride = (size_t)gridDim.x * blockDim.x;
  for (; i < n; i += stride) out[i] = (__bf16)in[i];
}

// relu + f32 -> bf16
__global__ __launch_bounds__(256) void relu_cvt_bf16(const float* __restrict__ in,
                                                     __bf16* __restrict__ out,
                                                     size_t n) {
  size_t i = (size_t)blockIdx.x * blockDim.x + threadIdx.x;
  size_t stride = (size_t)gridDim.x * blockDim.x;
  for (; i < n; i += stride) out[i] = (__bf16)fmaxf(in[i], 0.0f);
}

// zero fill f32 (accumulators must start at 0 each launch)
__global__ __launch_bounds__(256) void zero_f32(float* __restrict__ p, size_t n) {
  size_t i = (size_t)blockIdx.x * blockDim.x + threadIdx.x;
  size_t stride = (size_t)gridDim.x * blockDim.x;
  for (; i < n; i += stride) p[i] = 0.0f;
}

// ---------------------------------------------------------------------------
// WMMA bf16 GEMM:  C[M,N] = A[M,K] * Bt[N,K]^T + bias[N]
//   A row-major bf16, Bt row-major bf16 (torch Linear weight layout [out,in]),
//   C row-major f32.  K multiple of 32, N multiple of 128*gridDim.y... per-block
//   tile = 64 rows x 128 cols; wave tile = 16 rows x 64 cols (4 wmma accums).
// ---------------------------------------------------------------------------
__device__ inline v16bf load_a_frag(const __bf16* __restrict__ A, int lda,
                                    int m0, int k0, int lane, int M) {
  const int half = lane >> 4;          // 0: K 0..15 ; 1: K 16..31
  int row = m0 + (lane & 15);
  if (row >= M) row = M - 1;           // clamp (store is guarded instead)
  const __bf16* p = A + (size_t)row * lda + k0 + half * 16;
  union { v16bf v; v8bf h[2]; } u;
  u.h[0] = *(const v8bf*)(p);          // K = base .. base+7
  u.h[1] = *(const v8bf*)(p + 8);      // K = base+8 .. base+15
  return u.v;
}

__device__ inline v16bf load_b_frag(const __bf16* __restrict__ Bt, int ldb,
                                    int n0, int k0, int lane) {
  const int half = lane >> 4;          // 0: K 0..15 ; 1: K 16..31
  const int col = n0 + (lane & 15);
  const __bf16* p = Bt + (size_t)col * ldb + k0 + half * 16;
  union { v16bf v; v8bf h[2]; } u;
  u.h[0] = *(const v8bf*)(p);
  u.h[1] = *(const v8bf*)(p + 8);
  return u.v;
}

__global__ __launch_bounds__(256) void gemm_bf16_wmma(
    const __bf16* __restrict__ A,   // [M,K]
    const __bf16* __restrict__ Bt,  // [N,K]
    const float*  __restrict__ bias,// [N]
    float* __restrict__ C,          // [M,N]
    int M, int N, int K) {
  const int tid  = threadIdx.x;
  const int wave = tid >> 5;   // 0..7
  const int lane = tid & 31;
  const int wm   = wave >> 1;  // 0..3  (M direction)
  const int wn   = wave & 1;   // 0..1  (N direction)

  const int m0 = blockIdx.x * 64 + wm * 16;
  const int n0 = blockIdx.y * 128 + wn * 64;

  v8f acc0 = {}, acc1 = {}, acc2 = {}, acc3 = {};

  for (int k0 = 0; k0 < K; k0 += 32) {
    v16bf a  = load_a_frag(A, K, m0, k0, lane, M);
    v16bf b0 = load_b_frag(Bt, K, n0 +  0, k0, lane);
    v16bf b1 = load_b_frag(Bt, K, n0 + 16, k0, lane);
    v16bf b2 = load_b_frag(Bt, K, n0 + 32, k0, lane);
    v16bf b3 = load_b_frag(Bt, K, n0 + 48, k0, lane);
    acc0 = __builtin_amdgcn_wmma_f32_16x16x32_bf16(false, a, false, b0, (short)0, acc0, false, false);
    acc1 = __builtin_amdgcn_wmma_f32_16x16x32_bf16(false, a, false, b1, (short)0, acc1, false, false);
    acc2 = __builtin_amdgcn_wmma_f32_16x16x32_bf16(false, a, false, b2, (short)0, acc2, false, false);
    acc3 = __builtin_amdgcn_wmma_f32_16x16x32_bf16(false, a, false, b3, (short)0, acc3, false, false);
  }

  // C/D layout: lane (0..15 | 16..31) -> N = n + (lane&15); VGPR r -> M = r | 8+r
  const int half   = lane >> 4;
  const int n_lane = lane & 15;
  const v8f accs[4] = {acc0, acc1, acc2, acc3};
#pragma unroll
  for (int j = 0; j < 4; ++j) {
    const int n  = n0 + j * 16 + n_lane;
    const float bv = bias[n];
#pragma unroll
    for (int r = 0; r < 8; ++r) {
      const int row = m0 + half * 8 + r;
      if (row < M) C[(size_t)row * N + n] = accs[j][r] + bv;
    }
  }
}

// ---------------------------------------------------------------------------
// SpMM scatter: out[src[e], :] += w[e] * sup[dst[e], :]   (one wave per edge)
// ---------------------------------------------------------------------------
template <int F>
__global__ __launch_bounds__(256) void spmm_atomic(
    const float* __restrict__ sup,
    const float* __restrict__ ew,
    const int*   __restrict__ esrc,
    const int*   __restrict__ edst,
    float* __restrict__ out, int E) {
  const int gwave = (int)((blockIdx.x * (size_t)blockDim.x + threadIdx.x) >> 5);
  const int lane  = threadIdx.x & 31;
  if (gwave >= E) return;
  const float w  = ew[gwave];
  const size_t s = (size_t)esrc[gwave] * F;
  const size_t d = (size_t)edst[gwave] * F;
#pragma unroll
  for (int f = lane; f < F; f += 32) {
    atomicAdd(&out[s + f], w * sup[d + f]);
  }
}

// ---------------------------------------------------------------------------
// launch
// ---------------------------------------------------------------------------
extern "C" void kernel_launch(void* const* d_in, const int* in_sizes, int n_in,
                              void* d_out, int out_size, void* d_ws, size_t ws_size,
                              hipStream_t stream) {
  const float* x    = (const float*)d_in[0];   // [N, 512]
  const float* ew   = (const float*)d_in[1];   // [E]
  const float* W1   = (const float*)d_in[2];   // [512, 512]
  const float* b1   = (const float*)d_in[3];   // [512]
  const float* W2   = (const float*)d_in[4];   // [128, 512]
  const float* b2   = (const float*)d_in[5];   // [128]
  const int*   esrc = (const int*)d_in[6];     // [E]
  const int*   edst = (const int*)d_in[7];     // [E]
  float* out = (float*)d_out;                  // [N, 128]

  // ---- workspace layout (reused across phases) ----
  char* ws = (char*)d_ws;
  const size_t xb_bytes  = (size_t)GCN_N * GCN_NFEAT * 2;  // 51,200,000
  const size_t w1b_bytes = (size_t)GCN_NHID * GCN_NFEAT * 2;
  const size_t w2b_bytes = (size_t)GCN_NCLASS * GCN_NHID * 2;
  __bf16* xb  = (__bf16*)ws;                       // phase1 A; reused as hb in phase2
  __bf16* w1b = (__bf16*)(ws + xb_bytes);
  __bf16* w2b = (__bf16*)(ws + xb_bytes + w1b_bytes);
  float*  sup = (float*)(ws + xb_bytes + w1b_bytes + w2b_bytes);  // sup1, later sup2
  float*  hacc = sup + (size_t)GCN_N * GCN_NHID;   // layer-1 scatter accumulator
  __bf16* hb = xb;                                 // relu'd hidden, bf16 (reuses xb)
  float*  sup2 = sup;                              // layer-2 dense result (reuses sup)

  const size_t n_xh = (size_t)GCN_N * GCN_NHID;    // 25.6M elems

  // 1) conversions to bf16
  cvt_f32_bf16<<<2048, 256, 0, stream>>>(x, xb, (size_t)GCN_N * GCN_NFEAT);
  cvt_f32_bf16<<<512, 256, 0, stream>>>(W1, w1b, (size_t)GCN_NHID * GCN_NFEAT);
  cvt_f32_bf16<<<128, 256, 0, stream>>>(W2, w2b, (size_t)GCN_NCLASS * GCN_NHID);

  // 2) zero the layer-1 accumulator
  zero_f32<<<2048, 256, 0, stream>>>(hacc, n_xh);

  // 3) GEMM1: sup = xb @ W1^T + b1   [50000, 512]
  {
    dim3 grid((GCN_N + 63) / 64, GCN_NHID / 128);
    gemm_bf16_wmma<<<grid, 256, 0, stream>>>(xb, w1b, b1, sup,
                                             GCN_N, GCN_NHID, GCN_NFEAT);
  }

  // 4) SpMM1: hacc[src] += w * sup[dst]   (one wave per edge)
  spmm_atomic<GCN_NHID><<<(GCN_E + 7) / 8, 256, 0, stream>>>(sup, ew, esrc, edst,
                                                             hacc, GCN_E);

  // 5) relu + convert to bf16 (hb overwrites xb, no longer needed)
  relu_cvt_bf16<<<2048, 256, 0, stream>>>(hacc, hb, n_xh);

  // 6) zero output accumulator
  zero_f32<<<2048, 256, 0, stream>>>(out, (size_t)GCN_N * GCN_NCLASS);

  // 7) GEMM2: sup2 = hb @ W2^T + b2   [50000, 128]  (overwrites sup1 region)
  {
    dim3 grid((GCN_N + 63) / 64, GCN_NCLASS / 128);
    gemm_bf16_wmma<<<grid, 256, 0, stream>>>(hb, w2b, b2, sup2,
                                             GCN_N, GCN_NCLASS, GCN_NHID);
  }

  // 8) SpMM2: out[src] += w * sup2[dst]
  spmm_atomic<GCN_NCLASS><<<(GCN_E + 7) / 8, 256, 0, stream>>>(sup2, ew, esrc, edst,
                                                               out, GCN_E);
}
